// DoubleStreamBlock_10831907520732
// MI455X (gfx1250) — matrix operand order repaired and checked by reading, not compiled, not run on hardware
//
#include <hip/hip_runtime.h>
#include <hip/hip_bf16.h>

// ---------------------------------------------------------------------------
// DoubleStreamBlock forward for MI455X (gfx1250, wave32, WMMA).
// Heavy matmuls: v_wmma_f32_16x16x32_bf16 (f32 accumulate).
// GEMM data movement: async global->LDS DMA (ASYNCcnt) with double buffering.
// ---------------------------------------------------------------------------

#define D_MODEL 1024
#define N_HEADS 16
#define HD      64
#define TXT_LEN 512
#define IMG_LEN 2048
#define SEQ_LEN 2560
#define MLP_DIM 4096

#ifndef __has_builtin
#define __has_builtin(x) 0
#endif
#if __has_builtin(__builtin_amdgcn_global_load_async_to_lds_b128) && \
    __has_builtin(__builtin_amdgcn_s_wait_asynccnt)
#define HAVE_ASYNC_LDS 1
#else
#define HAVE_ASYNC_LDS 0
#endif

#define GLOBAL_AS __attribute__((address_space(1)))
#define LDS_AS    __attribute__((address_space(3)))

typedef int v4i_vs __attribute__((vector_size(4 * sizeof(int))));  // builtin's b128 type

typedef __attribute__((ext_vector_type(16))) __bf16 v16bf;
typedef __attribute__((ext_vector_type(8)))  float  v8f;

struct FragU { unsigned int u[8]; };   // 8 VGPRs = 16 bf16 = one WMMA A/B operand

__device__ __forceinline__ v8f wmma_bf16(const FragU& a, const FragU& b, v8f c) {
  return __builtin_amdgcn_wmma_f32_16x16x32_bf16(
      false, __builtin_bit_cast(v16bf, a),
      false, __builtin_bit_cast(v16bf, b),
      (short)0, c, false, false);
}

__device__ __forceinline__ unsigned short f2bf(float f) {
  unsigned int u = __float_as_uint(f);
  u += 0x7fffu + ((u >> 16) & 1u);          // round-to-nearest-even
  return (unsigned short)(u >> 16);
}
__device__ __forceinline__ unsigned int packbf(float a, float b) {
  return (unsigned int)f2bf(a) | ((unsigned int)f2bf(b) << 16);
}

// --------------------------- elementwise cast ------------------------------
__global__ void cast_f32_bf16(const float* __restrict__ in,
                              unsigned short* __restrict__ out, int n) {
  int i = blockIdx.x * blockDim.x + threadIdx.x;
  if (i < n) out[i] = f2bf(in[i]);
}

// --------------------- modulation GEMV: silu(vec) @ W + b ------------------
// out: 6144 floats = [shift1, scale1, gate1, shift2, scale2, gate2] x 1024
__global__ void mod_gemv(const float* __restrict__ vec, const float* __restrict__ W,
                         const float* __restrict__ b, float* __restrict__ out) {
  __shared__ float sv[D_MODEL];
  int tid = threadIdx.x;
  for (int k = tid; k < D_MODEL; k += 256) {
    float x = vec[k];
    sv[k] = x / (1.0f + __expf(-x));
  }
  __syncthreads();
  int j = blockIdx.x * 256 + tid;
  float acc = 0.f;
  for (int k = 0; k < D_MODEL; ++k)
    acc += sv[k] * W[(size_t)k * (6 * D_MODEL) + j];
  out[j] = acc + b[j];
}

// ------------- LayerNorm + (1+scale)*ln + shift, output bf16 ---------------
__global__ void ln_mod_cast(const float* __restrict__ x, const float* __restrict__ s,
                            const float* __restrict__ b, const float* __restrict__ shift,
                            const float* __restrict__ scale,
                            unsigned short* __restrict__ out) {
  int row = blockIdx.x, tid = threadIdx.x;
  const float* xr = x + (size_t)row * D_MODEL;
  float s1 = 0.f, s2 = 0.f;
  float v[4];
#pragma unroll
  for (int i = 0; i < 4; ++i) {
    float t = xr[tid + i * 256];
    v[i] = t; s1 += t; s2 += t * t;
  }
#pragma unroll
  for (int off = 1; off < 32; off <<= 1) {
    s1 += __shfl_xor(s1, off, 32);
    s2 += __shfl_xor(s2, off, 32);
  }
  __shared__ float ws1[8], ws2[8];
  int wid = tid >> 5;
  if ((tid & 31) == 0) { ws1[wid] = s1; ws2[wid] = s2; }
  __syncthreads();
  float S1 = 0.f, S2 = 0.f;
#pragma unroll
  for (int i = 0; i < 8; ++i) { S1 += ws1[i]; S2 += ws2[i]; }
  float mu  = S1 * (1.0f / D_MODEL);
  float var = S2 * (1.0f / D_MODEL) - mu * mu;
  float inv = rsqrtf(var + 1e-6f);
#pragma unroll
  for (int i = 0; i < 4; ++i) {
    int c = tid + i * 256;
    float ln = (v[i] - mu) * inv * s[c] + b[c];
    float xm = (1.0f + scale[c]) * ln + shift[c];
    out[(size_t)row * D_MODEL + c] = f2bf(xm);
  }
}

// ----------------------------- bf16 WMMA GEMM ------------------------------
// C(MxN) = A(MxK,bf16) @ B(KxN,bf16), epilogues:
//   mode 0: outf = acc + bias
//   mode 1: outf = residual + gate * (acc + bias)
//   mode 2: outbf = gelu_tanh(acc + bias)   (bf16 store, feeds next GEMM)
// Block tile 128x64, BK=32, 8 waves (4Mx2N), 32x32 per wave = 4 WMMA/K-step.
// Tiles streamed global->LDS with async DMA, double buffered.
__global__ __launch_bounds__(256) void gemm_bf16(
    const unsigned short* __restrict__ A, const unsigned short* __restrict__ B,
    const float* __restrict__ bias, float* __restrict__ outf,
    unsigned short* __restrict__ outbf, const float* __restrict__ residual,
    const float* __restrict__ gate, int M, int N, int K, int mode) {
  __shared__ unsigned int aT[2][128][16];   // 2 x (128 rows x 32 bf16)
  __shared__ unsigned int bT[2][32][32];    // 2 x (32 K-rows x 64 bf16)
  int tid  = threadIdx.x;
  int lane = tid & 31, wid = tid >> 5;
  int waveM = wid & 3, waveN = wid >> 2;
  int half = lane >> 4, l16 = lane & 15;
  int bM = blockIdx.y * 128, bN = blockIdx.x * 64;
  const unsigned int* Au = (const unsigned int*)A;
  const unsigned int* Bu = (const unsigned int*)B;
  int Ks2 = K >> 1, Ns2 = N >> 1;

  // per-thread streaming slots (pointer-increment addressing, no per-iter mul)
  int ar0 = tid >> 2, ac0 = (tid & 3) * 4;   // A piece 0: rows 0..63
  int ar1 = ar0 + 64, ac1 = ac0;             // A piece 1: rows 64..127
  int br  = tid >> 3, bc = (tid & 7) * 4;    // B: 32 rows x 8 uint4
  const unsigned int* aG0 = Au + (size_t)(bM + ar0) * Ks2 + ac0;
  const unsigned int* aG1 = Au + (size_t)(bM + ar1) * Ks2 + ac1;
  const unsigned int* bG  = Bu + (size_t)br * Ns2 + (bN >> 1) + bc;
  const size_t bStep = (size_t)32 * Ns2;

  v8f zero = {};
  v8f acc[2][2] = {{zero, zero}, {zero, zero}};

  auto compute = [&](int bufi) {
    FragU af[2], bfr[2];
#pragma unroll
    for (int ms = 0; ms < 2; ++ms) {        // A frag: lane = row, K split per half
      int row = waveM * 32 + ms * 16 + l16;
#pragma unroll
      for (int v = 0; v < 8; ++v)
        af[ms].u[v] = aT[bufi][row][(v & 3) + 8 * (v >> 2) + 4 * half];
    }
#pragma unroll
    for (int ns = 0; ns < 2; ++ns) {        // B frag: lane = N column, K split per half
#pragma unroll
      for (int v = 0; v < 8; ++v) {
        int kk = 2 * ((v & 3) + 8 * (v >> 2) + 4 * half);
        int c = waveN * 16 + ns * 8 + (l16 >> 1), sh = (l16 & 1) * 16;
        bfr[ns].u[v] = ((bT[bufi][kk][c] >> sh) & 0xffffu) | ((bT[bufi][kk + 1][c] >> sh) << 16);
      }
    }
#pragma unroll
    for (int ms = 0; ms < 2; ++ms)
#pragma unroll
      for (int ns = 0; ns < 2; ++ns)
        acc[ms][ns] = wmma_bf16(af[ms], bfr[ns], acc[ms][ns]);
  };

#if HAVE_ASYNC_LDS
  // --- async DMA path: GLOBAL_LOAD_ASYNC_TO_LDS_B128, ASYNCcnt tracked ---
#define ISSUE_TILE(bufi)                                                        \
  do {                                                                          \
    __builtin_amdgcn_global_load_async_to_lds_b128(                             \
        (GLOBAL_AS v4i_vs*)aG0, (LDS_AS v4i_vs*)&aT[bufi][ar0][ac0], 0, 0);     \
    __builtin_amdgcn_global_load_async_to_lds_b128(                             \
        (GLOBAL_AS v4i_vs*)aG1, (LDS_AS v4i_vs*)&aT[bufi][ar1][ac1], 0, 0);     \
    __builtin_amdgcn_global_load_async_to_lds_b128(                             \
        (GLOBAL_AS v4i_vs*)bG,  (LDS_AS v4i_vs*)&bT[bufi][br][bc], 0, 0);       \
    aG0 += 16; aG1 += 16; bG += bStep;                                          \
  } while (0)

  ISSUE_TILE(0);
  int buf = 0;
  for (int kt = 0; kt < K; kt += 32) {
    if (kt + 32 < K) {
      ISSUE_TILE(buf ^ 1);                       // next tile in flight
      __builtin_amdgcn_s_wait_asynccnt(3);       // current tile landed
    } else {
      __builtin_amdgcn_s_wait_asynccnt(0);
    }
    __syncthreads();
    compute(buf);
    buf ^= 1;
  }
#undef ISSUE_TILE
#else
  // --- fallback: register-staged, double buffered, loads overlap compute ---
  uint4 rA0, rA1, rB;
#define LOADR()                                                                 \
  do {                                                                          \
    rA0 = *(const uint4*)aG0; rA1 = *(const uint4*)aG1; rB = *(const uint4*)bG; \
    aG0 += 16; aG1 += 16; bG += bStep;                                          \
  } while (0)
#define STORER(bufi)                                                            \
  do {                                                                          \
    *(uint4*)&aT[bufi][ar0][ac0] = rA0;                                         \
    *(uint4*)&aT[bufi][ar1][ac1] = rA1;                                         \
    *(uint4*)&bT[bufi][br][bc]   = rB;                                          \
  } while (0)
  LOADR();
  STORER(0);
  __syncthreads();
  int buf = 0;
  for (int kt = 0; kt < K; kt += 32) {
    bool next = (kt + 32) < K;
    if (next) {
      __builtin_prefetch(aG0 + 16, 0, 3);
      __builtin_prefetch(bG + bStep, 0, 3);
      LOADR();                                   // next tile into regs
    }
    compute(buf);                                // overlaps with loads
    if (next) STORER(buf ^ 1);
    __syncthreads();
    buf ^= 1;
  }
#undef LOADR
#undef STORER
#endif

  // epilogue
#pragma unroll
  for (int ms = 0; ms < 2; ++ms) {
#pragma unroll
    for (int ns = 0; ns < 2; ++ns) {
      int colg = bN + waveN * 32 + ns * 16 + l16;
      float bb = bias[colg];
#pragma unroll
      for (int r = 0; r < 8; ++r) {
        int rowg = bM + waveM * 32 + ms * 16 + r + 8 * half;
        float val = acc[ms][ns][r] + bb;
        size_t o = (size_t)rowg * N + colg;
        if (mode == 0) {
          outf[o] = val;
        } else if (mode == 1) {
          outf[o] = residual[o] + gate[colg] * val;
        } else {
          float x3 = val * val * val;
          float g = 0.5f * val * (1.0f + tanhf(0.7978845608f * (val + 0.044715f * x3)));
          outbf[o] = f2bf(g);
        }
      }
    }
  }
}

// ---------------- QKV post: RMS norm + RoPE, pack bf16 tiles ---------------
// q,v: [H][L][64] bf16 ; k stored transposed: [H][64][L] bf16 (B-frag friendly)
__global__ void qkv_post(const float* __restrict__ qkv_txt, const float* __restrict__ qkv_img,
                         const float* __restrict__ pe,
                         const float* __restrict__ qs_txt, const float* __restrict__ ks_txt,
                         const float* __restrict__ qs_img, const float* __restrict__ ks_img,
                         unsigned int* __restrict__ qbuf, unsigned short* __restrict__ kT,
                         unsigned int* __restrict__ vbuf) {
  int tid = threadIdx.x;
  int lane = tid & 31;
  int wg = blockIdx.x * 8 + (tid >> 5);       // one wave per (l, h)
  int l = wg >> 4, h = wg & 15;
  const float *row, *qs, *ks;
  if (l < TXT_LEN) { row = qkv_txt + (size_t)l * (3 * D_MODEL); qs = qs_txt; ks = ks_txt; }
  else             { row = qkv_img + (size_t)(l - TXT_LEN) * (3 * D_MODEL); qs = qs_img; ks = ks_img; }
  int d0 = 2 * lane, d1 = d0 + 1;
  float qx = row[h * HD + d0],               qy = row[h * HD + d1];
  float kx = row[D_MODEL + h * HD + d0],     ky = row[D_MODEL + h * HD + d1];
  float vx = row[2 * D_MODEL + h * HD + d0], vy = row[2 * D_MODEL + h * HD + d1];
  float qss = qx * qx + qy * qy;
  float kss = kx * kx + ky * ky;
#pragma unroll
  for (int off = 1; off < 32; off <<= 1) {
    qss += __shfl_xor(qss, off, 32);
    kss += __shfl_xor(kss, off, 32);
  }
  float qr = rsqrtf(qss * (1.0f / HD) + 1e-6f);
  float kr = rsqrtf(kss * (1.0f / HD) + 1e-6f);
  qx *= qr * qs[d0]; qy *= qr * qs[d1];
  kx *= kr * ks[d0]; ky *= kr * ks[d1];
  const float* p = pe + (size_t)l * 128 + lane * 4;   // pe[l][i][j][m]
  float e00 = p[0], e01 = p[1], e10 = p[2], e11 = p[3];
  float qe = e00 * qx + e01 * qy, qo = e10 * qx + e11 * qy;
  float ke = e00 * kx + e01 * ky, ko = e10 * kx + e11 * ky;
  size_t base = ((size_t)h * SEQ_LEN + l) * 32 + lane;
  qbuf[base] = packbf(qe, qo);
  vbuf[base] = packbf(vx, vy);
  kT[(size_t)(h * HD + d0) * SEQ_LEN + l] = f2bf(ke);
  kT[(size_t)(h * HD + d1) * SEQ_LEN + l] = f2bf(ko);
}

// ------------------------ flash attention (WMMA) ---------------------------
// 4 waves/block; each wave owns a 16-query tile of one head, streams keys in
// 32-chunks with online softmax. 8 WMMAs per chunk (4 QK^T + 4 PV).
// P round-trips through a private per-wave LDS tile (in-order LDS pipe, so a
// wave-level ordering fence suffices -- no cross-wave s_barrier needed).
__global__ __launch_bounds__(128) void flash_attn(
    const unsigned int* __restrict__ qbuf, const unsigned short* __restrict__ kT,
    const unsigned int* __restrict__ vbuf, float* __restrict__ attn) {
  __shared__ unsigned short pbuf[4][16][32];
  int tid = threadIdx.x;
  int lane = tid & 31, w = tid >> 5;
  int half = lane >> 4, l16 = lane & 15;
  int h  = blockIdx.x;
  int qt = blockIdx.y * 4 + w;                // 16-row query tile index
  FragU qa[2];
  {
    size_t qb = ((size_t)h * SEQ_LEN + qt * 16 + l16) * 32;
#pragma unroll
    for (int f = 0; f < 2; ++f)
#pragma unroll
      for (int v = 0; v < 8; ++v)
        qa[f].u[v] = qbuf[qb + f * 16 + (v & 3) + 8 * (v >> 2) + 4 * half];
  }
  const unsigned int* kTu = (const unsigned int*)kT;
  float mst[8], lst[8];
  v8f zero = {};
  v8f o[4] = {zero, zero, zero, zero};
#pragma unroll
  for (int r = 0; r < 8; ++r) { mst[r] = -3.0e38f; lst[r] = 0.f; }
  unsigned int* pu = (unsigned int*)&pbuf[w][0][0];
  for (int kc = 0; kc < SEQ_LEN; kc += 32) {
    v8f s[2] = {zero, zero};
#pragma unroll
    for (int t = 0; t < 2; ++t) {             // 16-key subtiles
#pragma unroll
      for (int f = 0; f < 2; ++f) {           // d-dim split 2 x 32
        FragU kb;
        int key = kc + t * 16 + l16;          // B frag: lane = N(key), K(dim) split per half
        size_t kbase = (size_t)(h * HD + f * 32) * SEQ_LEN + key;
#pragma unroll
        for (int v = 0; v < 8; ++v) {
          int dd = 2 * ((v & 3) + 8 * (v >> 2) + 4 * half);
          kb.u[v] = (unsigned int)kT[kbase + (size_t)dd * SEQ_LEN] | ((unsigned int)kT[kbase + (size_t)(dd + 1) * SEQ_LEN] << 16);
        }
        s[t] = wmma_bf16(qa[f], kb, s[t]);
      }
    }
#pragma unroll
    for (int r = 0; r < 8; ++r) {             // online softmax per row slot
      float s0 = s[0][r] * 0.125f;            // HD^-0.5
      float s1 = s[1][r] * 0.125f;
      float mx = fmaxf(s0, s1);
#pragma unroll
      for (int off = 1; off < 16; off <<= 1) mx = fmaxf(mx, __shfl_xor(mx, off, 32));
      float nm = fmaxf(mst[r], mx);
      float corr = __expf(mst[r] - nm);
      float p0 = __expf(s0 - nm);
      float p1 = __expf(s1 - nm);
      float rs = p0 + p1;
#pragma unroll
      for (int off = 1; off < 16; off <<= 1) rs += __shfl_xor(rs, off, 32);
      lst[r] = lst[r] * corr + rs;
      mst[r] = nm;
#pragma unroll
      for (int ns = 0; ns < 4; ++ns) o[ns][r] *= corr;
      pbuf[w][r + 8 * half][l16]      = f2bf(p0);
      pbuf[w][r + 8 * half][l16 + 16] = f2bf(p1);
    }
    __builtin_amdgcn_wave_barrier();           // order LDS store->load in wave
    FragU pa;                                  // P as A-fragment (16x32)
#pragma unroll
    for (int v = 0; v < 8; ++v)
      pa.u[v] = pu[l16 * 16 + (v & 3) + 8 * (v >> 2) + 4 * half];
#pragma unroll
    for (int ns = 0; ns < 4; ++ns) {           // O += P @ V (dims split 4 x 16)
      FragU vb;
      size_t vbase = ((size_t)h * SEQ_LEN + kc) * 32 + ns * 8 + (l16 >> 1);
      int sh = (l16 & 1) * 16;
#pragma unroll
      for (int v = 0; v < 8; ++v) {
        int kk = 2 * ((v & 3) + 8 * (v >> 2) + 4 * half);
        vb.u[v] = ((vbuf[vbase + (size_t)kk * 32] >> sh) & 0xffffu) | ((vbuf[vbase + (size_t)(kk + 1) * 32] >> sh) << 16);
      }
      o[ns] = wmma_bf16(pa, vb, o[ns]);
    }
    __builtin_amdgcn_wave_barrier();
  }
#pragma unroll
  for (int r = 0; r < 8; ++r) {
    float inv = 1.0f / lst[r];
    int lq = qt * 16 + r + 8 * half;
#pragma unroll
    for (int ns = 0; ns < 4; ++ns) {
      int col = h * HD + ns * 16 + l16;        // (L, H*HD) head-major layout
      attn[(size_t)lq * D_MODEL + col] = o[ns][r] * inv;
    }
  }
}

// ------------------------------ launcher -----------------------------------
extern "C" void kernel_launch(void* const* d_in, const int* in_sizes, int n_in,
                              void* d_out, int out_size, void* d_ws, size_t ws_size,
                              hipStream_t stream) {
  (void)in_sizes; (void)n_in; (void)out_size; (void)ws_size;
  const float* img = (const float*)d_in[0];
  const float* txt = (const float*)d_in[1];
  const float* vec = (const float*)d_in[2];
  const float* pe  = (const float*)d_in[3];
  // per-stream params: 0 mod_w 1 mod_b 2 n1s 3 n1b 4 qkv_w 5 qkv_b 6 qsc 7 ksc
  //                    8 proj_w 9 proj_b 10 n2s 11 n2b 12 w1 13 b1 14 w2 15 b2
  const float* P[2][16];
  for (int s = 0; s < 2; ++s)
    for (int i = 0; i < 16; ++i)
      P[s][i] = (const float*)d_in[4 + s * 16 + i];   // s=0 img, s=1 txt

  char* ws = (char*)d_ws;
  size_t off = 0;
  auto alloc = [&](size_t bytes) -> void* {
    void* p = ws + off;
    off += (bytes + 255) & ~(size_t)255;
    return p;
  };
  unsigned short* wb_qkv[2]  = {(unsigned short*)alloc((size_t)D_MODEL * 3 * D_MODEL * 2),
                                (unsigned short*)alloc((size_t)D_MODEL * 3 * D_MODEL * 2)};
  unsigned short* wb_proj[2] = {(unsigned short*)alloc((size_t)D_MODEL * D_MODEL * 2),
                                (unsigned short*)alloc((size_t)D_MODEL * D_MODEL * 2)};
  unsigned short* wb_w1[2]   = {(unsigned short*)alloc((size_t)D_MODEL * MLP_DIM * 2),
                                (unsigned short*)alloc((size_t)D_MODEL * MLP_DIM * 2)};
  unsigned short* wb_w2[2]   = {(unsigned short*)alloc((size_t)MLP_DIM * D_MODEL * 2),
                                (unsigned short*)alloc((size_t)MLP_DIM * D_MODEL * 2)};
  float* modv[2] = {(float*)alloc(6 * D_MODEL * 4), (float*)alloc(6 * D_MODEL * 4)};
  unsigned short* xm_img = (unsigned short*)alloc((size_t)IMG_LEN * D_MODEL * 2);
  unsigned short* xm_txt = (unsigned short*)alloc((size_t)TXT_LEN * D_MODEL * 2);
  float* qkv_img = (float*)alloc((size_t)IMG_LEN * 3 * D_MODEL * 4);
  float* qkv_txt = (float*)alloc((size_t)TXT_LEN * 3 * D_MODEL * 4);
  unsigned int*   qbuf = (unsigned int*)alloc((size_t)N_HEADS * SEQ_LEN * HD * 2);
  unsigned short* kT   = (unsigned short*)alloc((size_t)N_HEADS * HD * SEQ_LEN * 2);
  unsigned int*   vbuf = (unsigned int*)alloc((size_t)N_HEADS * SEQ_LEN * HD * 2);
  float*          attn_f = (float*)alloc((size_t)SEQ_LEN * D_MODEL * 4);
  unsigned short* attn_b = (unsigned short*)alloc((size_t)SEQ_LEN * D_MODEL * 2);
  unsigned short* h_img  = (unsigned short*)alloc((size_t)IMG_LEN * D_MODEL * 2);
  unsigned short* h_txt  = (unsigned short*)alloc((size_t)TXT_LEN * D_MODEL * 2);
  unsigned short* m1_img = (unsigned short*)alloc((size_t)IMG_LEN * MLP_DIM * 2);
  unsigned short* m1_txt = (unsigned short*)alloc((size_t)TXT_LEN * MLP_DIM * 2);

  float* out_img = (float*)d_out;
  float* out_txt = out_img + (size_t)IMG_LEN * D_MODEL;

  auto cast = [&](const float* in, unsigned short* out_, int n) {
    cast_f32_bf16<<<(n + 255) / 256, 256, 0, stream>>>(in, out_, n);
  };
  // weight casts (one-time fp32 -> bf16)
  for (int s = 0; s < 2; ++s) {
    cast(P[s][4],  wb_qkv[s],  D_MODEL * 3 * D_MODEL);
    cast(P[s][8],  wb_proj[s], D_MODEL * D_MODEL);
    cast(P[s][12], wb_w1[s],   D_MODEL * MLP_DIM);
    cast(P[s][14], wb_w2[s],   MLP_DIM * D_MODEL);
  }
  // modulation vectors
  mod_gemv<<<24, 256, 0, stream>>>(vec, P[0][0], P[0][1], modv[0]);
  mod_gemv<<<24, 256, 0, stream>>>(vec, P[1][0], P[1][1], modv[1]);
  // LN1 + modulate -> bf16
  ln_mod_cast<<<IMG_LEN, 256, 0, stream>>>(img, P[0][2], P[0][3],
                                           modv[0] + 0, modv[0] + 1024, xm_img);
  ln_mod_cast<<<TXT_LEN, 256, 0, stream>>>(txt, P[1][2], P[1][3],
                                           modv[1] + 0, modv[1] + 1024, xm_txt);
  // QKV GEMMs
  gemm_bf16<<<dim3(3 * D_MODEL / 64, IMG_LEN / 128), 256, 0, stream>>>(
      xm_img, wb_qkv[0], P[0][5], qkv_img, nullptr, nullptr, nullptr,
      IMG_LEN, 3 * D_MODEL, D_MODEL, 0);
  gemm_bf16<<<dim3(3 * D_MODEL / 64, TXT_LEN / 128), 256, 0, stream>>>(
      xm_txt, wb_qkv[1], P[1][5], qkv_txt, nullptr, nullptr, nullptr,
      TXT_LEN, 3 * D_MODEL, D_MODEL, 0);
  // RMS + RoPE + pack (txt tokens first, then img, matching concat order)
  qkv_post<<<SEQ_LEN * N_HEADS / 8, 256, 0, stream>>>(
      qkv_txt, qkv_img, pe, P[1][6], P[1][7], P[0][6], P[0][7], qbuf, kT, vbuf);
  // attention
  flash_attn<<<dim3(N_HEADS, SEQ_LEN / 64), 128, 0, stream>>>(qbuf, kT, vbuf, attn_f);
  cast(attn_f, attn_b, SEQ_LEN * D_MODEL);
  // output projections: x = x + gate1 * (attn @ proj_w + proj_b)
  gemm_bf16<<<dim3(D_MODEL / 64, IMG_LEN / 128), 256, 0, stream>>>(
      attn_b + (size_t)TXT_LEN * D_MODEL, wb_proj[0], P[0][9], out_img, nullptr,
      img, modv[0] + 2048, IMG_LEN, D_MODEL, D_MODEL, 1);
  gemm_bf16<<<dim3(D_MODEL / 64, TXT_LEN / 128), 256, 0, stream>>>(
      attn_b, wb_proj[1], P[1][9], out_txt, nullptr,
      txt, modv[1] + 2048, TXT_LEN, D_MODEL, D_MODEL, 1);
  // MLP (img): LN2 -> GEMM+GELU(bf16) -> GEMM + residual*gate
  ln_mod_cast<<<IMG_LEN, 256, 0, stream>>>(out_img, P[0][10], P[0][11],
                                           modv[0] + 3072, modv[0] + 4096, h_img);
  gemm_bf16<<<dim3(MLP_DIM / 64, IMG_LEN / 128), 256, 0, stream>>>(
      h_img, wb_w1[0], P[0][13], nullptr, m1_img, nullptr, nullptr,
      IMG_LEN, MLP_DIM, D_MODEL, 2);
  gemm_bf16<<<dim3(D_MODEL / 64, IMG_LEN / 128), 256, 0, stream>>>(
      m1_img, wb_w2[0], P[0][15], out_img, nullptr, out_img, modv[0] + 5120,
      IMG_LEN, D_MODEL, MLP_DIM, 1);
  // MLP (txt)
  ln_mod_cast<<<TXT_LEN, 256, 0, stream>>>(out_txt, P[1][10], P[1][11],
                                           modv[1] + 3072, modv[1] + 4096, h_txt);
  gemm_bf16<<<dim3(MLP_DIM / 64, TXT_LEN / 128), 256, 0, stream>>>(
      h_txt, wb_w1[1], P[1][13], nullptr, m1_txt, nullptr, nullptr,
      TXT_LEN, MLP_DIM, D_MODEL, 2);
  gemm_bf16<<<dim3(D_MODEL / 64, TXT_LEN / 128), 256, 0, stream>>>(
      m1_txt, wb_w2[1], P[1][15], out_txt, nullptr, out_txt, modv[1] + 5120,
      TXT_LEN, D_MODEL, MLP_DIM, 1);
}
